// LaplaceProcessor_89343909692235
// MI455X (gfx1250) — matrix operand order repaired and checked by reading, not compiled, not run on hardware
//
#include <hip/hip_runtime.h>
#include <hip/hip_bf16.h>

typedef __attribute__((ext_vector_type(16))) _Float16 v16h;
typedef __attribute__((ext_vector_type(8)))  _Float16 v8h;
typedef __attribute__((ext_vector_type(8)))  float    v8f;

#define N_NODES 50000
#define N_EDGES 500000
#define D 128
#define K_PHI 288   // 2*D + E_DIM padded 258 -> 288 (multiple of 32)
#define K_GAM 256
#define ETILES 10   // 500000/16/10 = 3125 blocks
#define NTILES 5    // 50000/16/5   = 625  blocks
#define USE_ASYNC 1

// ============ weight convert into per-lane-swizzled WMMA B layout ============
// dst flat index o = (((l*QC + q)*8 + wave)*32 + lane)*16 + i
// value = W[K][c] with K = 32q + 16*(lane>=16) + i, c = 16*wave + (lane&15)
// fold!=0 (phi_W1): rows 0..127 become W1a - W1b (handles the h_j - h_i term
// algebraically: h_i@W1a + (h_j-h_i)@W1b == h_i@(W1a-W1b) + h_j@W1b)
__global__ void cvt_swz(const float* __restrict__ src, _Float16* __restrict__ dst,
                        int QC, int rows, int fold) {
    int o = blockIdx.x * blockDim.x + threadIdx.x;
    int total = 3 * QC * 4096;
    if (o >= total) return;
    int i    = o & 15;
    int lane = (o >> 4) & 31;
    int w    = (o >> 9) & 7;
    int q    = (o >> 12) % QC;
    int l    = o / (QC << 12);
    int K = (q << 5) + ((lane >> 4) << 4) + i;
    int c = (w << 4) + (lane & 15);
    const float* S = src + (size_t)l * rows * D;
    float v;
    if (fold && K < D)      v = S[K * D + c] - S[(K + D) * D + c];
    else if (K < rows)      v = S[K * D + c];
    else                    v = 0.f;
    dst[o] = (_Float16)v;
}

// ============ per-layer prep: snapshot h (f32 + f16), zero agg ============
__global__ void prep_layer(const float* __restrict__ src, float* __restrict__ hcur,
                           _Float16* __restrict__ h16, float* __restrict__ agg, int n) {
    int i = blockIdx.x * blockDim.x + threadIdx.x;
    if (i >= n) return;
    float v = src[i];
    hcur[i] = v;
    h16[i]  = (_Float16)v;
    agg[i]  = 0.f;
}

// 16-byte global->LDS segment copy (async DMA if available)
__device__ __forceinline__ void seg16(const _Float16* base, uint32_t elem_off,
                                      _Float16* lptr) {
#if USE_ASYNC
    uint32_t goff  = elem_off * 2u;
    uint32_t laddr = (uint32_t)(uintptr_t)lptr;
    asm volatile("global_load_async_to_lds_b128 %0, %1, %2"
                 :: "v"(laddr), "v"(goff), "s"(base) : "memory");
#else
    *(v8h*)lptr = *(const v8h*)(base + elem_off);
#endif
}
__device__ __forceinline__ void async_fence() {
#if USE_ASYNC
    asm volatile("s_wait_asynccnt 0x0" ::: "memory");
#endif
}

// load one swizzled B chunk: contiguous 32 bytes per lane -> 2x global_load_b128
__device__ __forceinline__ v16h loadB(const _Float16* W, int q, int wave, int lane) {
    const _Float16* wp = W + (((q << 3) + wave) << 9) + (lane << 4);
    v8h lo = *(const v8h*)wp;
    v8h hi = *(const v8h*)(wp + 8);
    return __builtin_shufflevector(lo, hi, 0,1,2,3,4,5,6,7,8,9,10,11,12,13,14,15);
}

// ============ edge message MLP + scatter-add (phi + segment_sum fused) ============
__global__ __launch_bounds__(256) void edge_mpnn(
    const _Float16* __restrict__ h16, const float* __restrict__ ea,
    const int* __restrict__ srcI, const int* __restrict__ dstI,
    const _Float16* __restrict__ W1, const float* __restrict__ b1,
    const _Float16* __restrict__ W2, const float* __restrict__ b2,
    float* __restrict__ agg)
{
    __shared__ _Float16 Alds[16 * K_PHI];
    __shared__ _Float16 Hlds[16 * D];

    const int tid  = threadIdx.x;
    const int lane = tid & 31;
    const int wave = tid >> 5;
    const int ncol = (wave << 4) | (lane & 15);
    const int k8   = (lane >> 4) << 3;
    const int m    = lane & 15;

    v16h B1[K_PHI / 32], B2[D / 32];
    for (int q = 0; q < K_PHI / 32; ++q) B1[q] = loadB(W1, q, wave, lane);
    for (int q = 0; q < D / 32; ++q)     B2[q] = loadB(W2, q, wave, lane);
    const float bias1 = b1[ncol];
    const float bias2 = b2[ncol];

    for (int t = 0; t < ETILES; ++t) {
        const int ebase = (blockIdx.x * ETILES + t) << 4;
        __syncthreads();  // previous tile fully consumed before LDS reuse

        // A tile [16, 288] = [h_dst | h_src | e | pad] via 512 async 16B segments
        for (int s = tid; s < 512; s += 256) {
            int r = s >> 5, seg = s & 31;
            int e = ebase + r;
            int node = (seg < 16) ? dstI[e] : srcI[e];
            int gcol = (seg & 15) << 3;
            seg16(h16, (uint32_t)(node * D + gcol), &Alds[r * K_PHI + (seg << 3)]);
        }
        {   // tail: cols 256..287 (edge_attr + zero pad), packed dword stores
            int r = tid >> 4, c2 = tid & 15;
            if (c2 == 0) {
                int e = ebase + r;
                Alds[r * K_PHI + 256] = (_Float16)ea[2 * e];
                Alds[r * K_PHI + 257] = (_Float16)ea[2 * e + 1];
            } else {
                *(uint32_t*)&Alds[r * K_PHI + 256 + (c2 << 1)] = 0u;
            }
        }
        async_fence();
        __syncthreads();

        // GEMM1: [16,288] x [288,16] -> relu -> f16 hidden tile
        v8f acc; for (int r = 0; r < 8; ++r) acc[r] = bias1;
        for (int q = 0; q < K_PHI / 32; ++q) {
            v16h a;
            for (int i = 0; i < 16; ++i) {
                int k = (k8 << 1) + i;   // contiguous k = 16*h + i, matches swizzled B
                a[i] = Alds[m * K_PHI + (q << 5) + k];
            }
            acc = __builtin_amdgcn_wmma_f32_16x16x32_f16(false, a, false, B1[q],
                                                         (short)0, acc, false, false);
        }
        for (int r = 0; r < 8; ++r)
            Hlds[(r + k8) * D + ncol] = (_Float16)fmaxf(acc[r], 0.f);
        __syncthreads();

        // GEMM2: [16,128] x [128,16] -> message -> atomic scatter on dst
        v8f acc2; for (int r = 0; r < 8; ++r) acc2[r] = bias2;
        for (int q = 0; q < D / 32; ++q) {
            v16h a;
            for (int i = 0; i < 16; ++i) {
                int k = (k8 << 1) + i;
                a[i] = Hlds[m * D + (q << 5) + k];
            }
            acc2 = __builtin_amdgcn_wmma_f32_16x16x32_f16(false, a, false, B2[q],
                                                          (short)0, acc2, false, false);
        }
        for (int r = 0; r < 8; ++r) {
            int d = dstI[ebase + r + k8];
            atomicAdd(agg + d * D + ncol, acc2[r]);
        }
    }
}

// ============ node update MLP + residual (gamma fused with h += ...) ============
__global__ __launch_bounds__(256) void node_mpnn(
    const _Float16* __restrict__ h16, const float* __restrict__ agg,
    const _Float16* __restrict__ G1, const float* __restrict__ gb1,
    const _Float16* __restrict__ G2, const float* __restrict__ gb2,
    const float* __restrict__ hin, float* __restrict__ out)
{
    __shared__ _Float16 Alds[16 * K_GAM];
    __shared__ _Float16 Hlds[16 * D];

    const int tid  = threadIdx.x;
    const int lane = tid & 31;
    const int wave = tid >> 5;
    const int ncol = (wave << 4) | (lane & 15);
    const int k8   = (lane >> 4) << 3;
    const int m    = lane & 15;

    v16h B1[K_GAM / 32], B2[D / 32];
    for (int q = 0; q < K_GAM / 32; ++q) B1[q] = loadB(G1, q, wave, lane);
    for (int q = 0; q < D / 32; ++q)     B2[q] = loadB(G2, q, wave, lane);
    const float bias1 = gb1[ncol];
    const float bias2 = gb2[ncol];

    for (int t = 0; t < NTILES; ++t) {
        const int nbase = (blockIdx.x * NTILES + t) << 4;
        __syncthreads();

        // A tile [16, 256] = [h | agg]: h via async 16B segments, agg via cvt+pack
        {
            int r = tid >> 4, seg = tid & 15;
            seg16(h16, (uint32_t)((nbase + r) * D + (seg << 3)),
                  &Alds[r * K_GAM + (seg << 3)]);
        }
        for (int p = tid; p < 16 * 64; p += 256) {
            int r = p >> 6, c2 = p & 63;
            float2 f = *(const float2*)(agg + (nbase + r) * D + (c2 << 1));
            union { _Float16 h[2]; uint32_t u; } u;
            u.h[0] = (_Float16)f.x; u.h[1] = (_Float16)f.y;
            *(uint32_t*)&Alds[r * K_GAM + D + (c2 << 1)] = u.u;
        }
        async_fence();
        __syncthreads();

        v8f acc; for (int r = 0; r < 8; ++r) acc[r] = bias1;
        for (int q = 0; q < K_GAM / 32; ++q) {
            v16h a;
            for (int i = 0; i < 16; ++i) {
                int k = (k8 << 1) + i;
                a[i] = Alds[m * K_GAM + (q << 5) + k];
            }
            acc = __builtin_amdgcn_wmma_f32_16x16x32_f16(false, a, false, B1[q],
                                                         (short)0, acc, false, false);
        }
        for (int r = 0; r < 8; ++r)
            Hlds[(r + k8) * D + ncol] = (_Float16)fmaxf(acc[r], 0.f);
        __syncthreads();

        v8f acc2; for (int r = 0; r < 8; ++r) acc2[r] = bias2;
        for (int q = 0; q < D / 32; ++q) {
            v16h a;
            for (int i = 0; i < 16; ++i) {
                int k = (k8 << 1) + i;
                a[i] = Hlds[m * D + (q << 5) + k];
            }
            acc2 = __builtin_amdgcn_wmma_f32_16x16x32_f16(false, a, false, B2[q],
                                                          (short)0, acc2, false, false);
        }
        for (int r = 0; r < 8; ++r) {
            int o = (nbase + r + k8) * D + ncol;
            out[o] = hin[o] + acc2[r];   // residual in fp32
        }
    }
}

extern "C" void kernel_launch(void* const* d_in, const int* in_sizes, int n_in,
                              void* d_out, int out_size, void* d_ws, size_t ws_size,
                              hipStream_t stream) {
    const float* h_in   = (const float*)d_in[0];
    const float* ea     = (const float*)d_in[1];
    const int*   eidx   = (const int*)  d_in[2];
    const float* phiW1  = (const float*)d_in[3];
    const float* phib1  = (const float*)d_in[4];
    const float* phiW2  = (const float*)d_in[5];
    const float* phib2  = (const float*)d_in[6];
    const float* gamW1  = (const float*)d_in[7];
    const float* gamb1  = (const float*)d_in[8];
    const float* gamW2  = (const float*)d_in[9];
    const float* gamb2  = (const float*)d_in[10];
    float* out = (float*)d_out;

    const int* srcI = eidx;
    const int* dstI = eidx + N_EDGES;

    char* ws = (char*)d_ws;
    size_t off = 0;
    auto take = [&](size_t bytes) { char* p = ws + off; off = (off + bytes + 255) & ~(size_t)255; return p; };
    _Float16* W1f = (_Float16*)take(3 * 9 * 4096 * 2);   // phi_W1 swizzled, folded
    _Float16* W2f = (_Float16*)take(3 * 4 * 4096 * 2);
    _Float16* G1f = (_Float16*)take(3 * 8 * 4096 * 2);
    _Float16* G2f = (_Float16*)take(3 * 4 * 4096 * 2);
    _Float16* h16 = (_Float16*)take((size_t)N_NODES * D * 2);
    float*    agg = (float*)   take((size_t)N_NODES * D * 4);
    float*    hcur= (float*)   take((size_t)N_NODES * D * 4);
    (void)ws_size; (void)in_sizes; (void)n_in; (void)out_size;

    cvt_swz<<<(3 * 9 * 4096 + 255) / 256, 256, 0, stream>>>(phiW1, W1f, 9, 2 * D + 2, 1);
    cvt_swz<<<(3 * 4 * 4096 + 255) / 256, 256, 0, stream>>>(phiW2, W2f, 4, D, 0);
    cvt_swz<<<(3 * 8 * 4096 + 255) / 256, 256, 0, stream>>>(gamW1, G1f, 8, K_GAM, 0);
    cvt_swz<<<(3 * 4 * 4096 + 255) / 256, 256, 0, stream>>>(gamW2, G2f, 4, D, 0);

    const int nd = N_NODES * D;
    const int edge_blocks = N_EDGES / (16 * ETILES);  // 3125
    const int node_blocks = N_NODES / (16 * NTILES);  // 625

    for (int l = 0; l < 3; ++l) {
        const float* src_h = (l == 0) ? h_in : hcur;
        float* layer_out = (l == 2) ? out : hcur;

        prep_layer<<<(nd + 255) / 256, 256, 0, stream>>>(src_h, hcur, h16, agg, nd);

        edge_mpnn<<<edge_blocks, 256, 0, stream>>>(
            h16, ea, srcI, dstI,
            W1f + (size_t)l * 9 * 4096, phib1 + l * D,
            W2f + (size_t)l * 4 * 4096, phib2 + l * D,
            agg);

        node_mpnn<<<node_blocks, 256, 0, stream>>>(
            h16, agg,
            G1f + (size_t)l * 8 * 4096, gamb1 + l * D,
            G2f + (size_t)l * 4 * 4096, gamb2 + l * D,
            hcur, layer_out);
    }
}
